// QCNN_simple_4bit_71141838291242
// MI455X (gfx1250) — 8-bit matrix operand order repaired and checked by reading, compiled, not run on hardware
//
#include <hip/hip_runtime.h>

typedef __attribute__((ext_vector_type(8))) int v8i;

#define OFF_POOL 256
#define OFF_W1P  8448
#define OFF_W2P  10496
#define OFF_XQ   32768
#define OFF_A1   4849664

__device__ __forceinline__ int imax(int a, int b) { return a > b ? a : b; }
__device__ __forceinline__ int imin(int a, int b) { return a < b ? a : b; }
__device__ __forceinline__ float ws_scale(const unsigned* u, int slot, float div) {
    return fmaxf(__uint_as_float(u[slot]), 1e-8f) / div;
}

__global__ void zero_ws(int* p, int n) {
    int i = blockIdx.x * 256 + threadIdx.x;
    if (i < n) p[i] = 0;
}

__global__ void absmax_kernel(const float* __restrict__ x, int n, unsigned* out) {
    __shared__ float red[256];
    int tid = threadIdx.x;
    float m = 0.f;
    for (int i = blockIdx.x * 256 + tid; i < n; i += gridDim.x * 256)
        m = fmaxf(m, fabsf(x[i]));
    red[tid] = m;
    __syncthreads();
    for (int s = 128; s > 0; s >>= 1) {
        if (tid < s) red[tid] = fmaxf(red[tid], red[tid + s]);
        __syncthreads();
    }
    if (tid == 0) atomicMax(out, __float_as_uint(red[0]));
}

__global__ void quantize_x(const float* __restrict__ x, const unsigned* __restrict__ u,
                           signed char* __restrict__ xq) {
    int idx = blockIdx.x * 256 + threadIdx.x;
    const int N = 32 * 3 * 224 * 224;
    if (idx >= N) return;
    float s_in = ws_scale(u, 0, 7.f);
    int b = idx / (3 * 224 * 224);
    int rem = idx % (3 * 224 * 224);
    int c = rem / (224 * 224);
    int hw = rem % (224 * 224);
    int h = hw / 224, w = hw % 224;
    float q = fminf(fmaxf(rintf(x[idx] / s_in), -8.f), 7.f);
    xq[((b * 224 + h) * 224 + w) * 3 + c] = (signed char)(int)q;
}

__global__ void pack_w1(const float* __restrict__ w1, const unsigned* __restrict__ u,
                        int* __restrict__ w1p) {
    int id = blockIdx.x * 256 + threadIdx.x;
    if (id >= 512) return;
    float sw = ws_scale(u, 1, 7.f);
    int t = id >> 8, lane = (id >> 3) & 31, v = id & 7;
    int oc = t * 16 + (lane & 15);
    unsigned d = 0;
    for (int bt = 0; bt < 4; bt++) {
        int K = ((v >= 4) ? 32 : 0) + ((lane >= 16) ? 16 : 0) + (v & 3) * 4 + bt;
        if (K < 27) {
            int tap = K / 3, ch = K - tap * 3, kr = tap / 3, kc = tap - kr * 3;
            float w = w1[((oc * 3 + ch) * 3 + kr) * 3 + kc];
            int q = (int)fminf(fmaxf(rintf(w / sw), -8.f), 7.f);
            d |= ((unsigned)(unsigned char)(signed char)q) << (bt * 8);
        }
    }
    w1p[id] = (int)d;
}

__global__ void pack_w2(const float* __restrict__ w2, const unsigned* __restrict__ u,
                        int* __restrict__ w2p) {
    int id = blockIdx.x * 256 + threadIdx.x;
    if (id >= 5120) return;
    float sw = ws_scale(u, 2, 7.f);
    int t = id / 1280, qc = (id / 256) % 5, lane = (id >> 3) & 31, v = id & 7;
    int oc = t * 16 + (lane & 15);
    unsigned d = 0;
    for (int bt = 0; bt < 4; bt++) {
        int K = qc * 64 + ((v >= 4) ? 32 : 0) + ((lane >= 16) ? 16 : 0) + (v & 3) * 4 + bt;
        if (K < 288) {
            int tap = K >> 5, ch = K & 31, kr = tap / 3, kc = tap - kr * 3;
            float w = w2[((oc * 32 + ch) * 3 + kr) * 3 + kc];
            int q = (int)fminf(fmaxf(rintf(w / sw), -8.f), 7.f);
            d |= ((unsigned)(unsigned char)(signed char)q) << (bt * 8);
        }
    }
    w2p[id] = (int)d;
}

__global__ __launch_bounds__(32) void conv1_wmma(
    const signed char* __restrict__ xq, const int* __restrict__ w1p,
    const float* __restrict__ b1, const unsigned* __restrict__ u,
    int* __restrict__ max1p, signed char* __restrict__ a1, int pass) {
    int lane = threadIdx.x;
    int m = lane & 15, hi = lane >> 4;
    int c0 = blockIdx.x << 4;
    int r = blockIdx.y;
    int bb = blockIdx.z;
    float s_in = ws_scale(u, 0, 7.f);
    float sw1 = ws_scale(u, 1, 7.f);
    float sconv = s_in * sw1;
    int cL = imin(c0 + m, 221);

    v8i A;
#pragma unroll
    for (int v = 0; v < 8; v++) {
        int Kb = ((v >> 2) << 5) + (hi << 4) + ((v & 3) << 2);
        unsigned d = 0;
#pragma unroll
        for (int bt = 0; bt < 4; bt++) {
            int K = Kb + bt;
            if (K < 27) {
                int tap = K / 3, ch = K - tap * 3, kr = tap / 3, kc = tap - kr * 3;
                int idx = ((bb * 224 + r + kr) * 224 + cL + kc) * 3 + ch;
                d |= ((unsigned)(unsigned char)xq[idx]) << (bt * 8);
            }
        }
        A[v] = (int)d;
    }

    const int4* wp = (const int4*)w1p;
    v8i acc[2];
#pragma unroll
    for (int t = 0; t < 2; t++) {
        int4 lo = wp[(t * 32 + lane) * 2 + 0];
        int4 h4 = wp[(t * 32 + lane) * 2 + 1];
        v8i B;
        B[0] = lo.x; B[1] = lo.y; B[2] = lo.z; B[3] = lo.w;
        B[4] = h4.x; B[5] = h4.y; B[6] = h4.z; B[7] = h4.w;
        v8i Z = (v8i)0;
        acc[t] = __builtin_amdgcn_wmma_i32_16x16x64_iu8(true, A, true, B, Z, false, false);
    }

    float s1 = 1.f;
    if (pass == 1) s1 = fmaxf(sconv * (float)imax(*max1p, 0), 1e-8f) / 15.f;

    int localmax = 0;
#pragma unroll
    for (int t = 0; t < 2; t++) {
        int oc = t * 16 + m;
        int bq = (int)rintf(b1[oc] / sconv);
#pragma unroll
        for (int i = 0; i < 8; i++) {
            int M = i + (hi << 3);
            int cc = c0 + M;
            int vv = acc[t][i] + bq;
            int rl = vv > 0 ? vv : 0;
            if (pass == 0) {
                if (cc < 222) localmax = imax(localmax, rl);
            } else if (cc < 222) {
                float q = rintf((float)rl * sconv / s1);
                int qi = imin(imax((int)q, 0), 15);
                a1[((bb * 222 + r) * 222 + cc) * 32 + oc] = (signed char)qi;
            }
        }
    }
    if (pass == 0) {
        for (int off = 16; off > 0; off >>= 1)
            localmax = imax(localmax, __shfl_xor(localmax, off, 32));
        if (lane == 0) atomicMax(max1p, localmax);
    }
}

__global__ __launch_bounds__(32) void conv2_wmma(
    const signed char* __restrict__ a1, const int* __restrict__ w2p,
    const float* __restrict__ b2, const unsigned* __restrict__ u,
    const int* __restrict__ max1p, int* __restrict__ max2p,
    int* __restrict__ pool, int pass) {
    int lane = threadIdx.x;
    int m = lane & 15, hi = lane >> 4;
    int c0 = blockIdx.x << 4;
    int r = blockIdx.y;
    int bb = blockIdx.z;
    float s_in = ws_scale(u, 0, 7.f);
    float sw1 = ws_scale(u, 1, 7.f);
    float sw2 = ws_scale(u, 2, 7.f);
    float s1 = fmaxf(s_in * sw1 * (float)imax(*max1p, 0), 1e-8f) / 15.f;
    float sconv = s1 * sw2;
    int cL = imin(c0 + m, 219);

    const int* a1i = (const int*)a1;
    const int4* wp = (const int4*)w2p;
    v8i acc[4];
#pragma unroll
    for (int t = 0; t < 4; t++) acc[t] = (v8i)0;

#pragma unroll
    for (int q = 0; q < 5; q++) {
        v8i A;
#pragma unroll
        for (int v = 0; v < 8; v++) {
            int Kb = q * 64 + ((v >> 2) << 5) + (hi << 4) + ((v & 3) << 2);
            int d = 0;
            if (Kb < 288) {
                int tap = Kb >> 5, ch = Kb & 31, kr = tap / 3, kc = tap - kr * 3;
                d = a1i[((bb * 222 + r + kr) * 222 + cL + kc) * 8 + (ch >> 2)];
            }
            A[v] = d;
        }
#pragma unroll
        for (int t = 0; t < 4; t++) {
            int4 lo = wp[((t * 5 + q) * 32 + lane) * 2 + 0];
            int4 h4 = wp[((t * 5 + q) * 32 + lane) * 2 + 1];
            v8i B;
            B[0] = lo.x; B[1] = lo.y; B[2] = lo.z; B[3] = lo.w;
            B[4] = h4.x; B[5] = h4.y; B[6] = h4.z; B[7] = h4.w;
            acc[t] = __builtin_amdgcn_wmma_i32_16x16x64_iu8(false, A, true, B, acc[t], false, false);
        }
    }

    float s2 = 1.f;
    if (pass == 1) s2 = fmaxf(sconv * (float)imax(*max2p, 0), 1e-8f) / 15.f;

    int localmax = 0;
#pragma unroll
    for (int t = 0; t < 4; t++) {
        int oc = t * 16 + m;
        int bq = (int)rintf(b2[oc] / sconv);
        int lanesum = 0;
#pragma unroll
        for (int i = 0; i < 8; i++) {
            int M = i + (hi << 3);
            int cc = c0 + M;
            int vv = acc[t][i] + bq;
            int rl = vv > 0 ? vv : 0;
            if (cc < 220) {
                if (pass == 0) {
                    localmax = imax(localmax, rl);
                } else {
                    float q = rintf((float)rl * sconv / s2);
                    lanesum += imin(imax((int)q, 0), 15);
                }
            }
        }
        if (pass == 1) atomicAdd(&pool[bb * 64 + oc], lanesum);
    }
    if (pass == 0) {
        for (int off = 16; off > 0; off >>= 1)
            localmax = imax(localmax, __shfl_xor(localmax, off, 32));
        if (lane == 0) atomicMax(max2p, localmax);
    }
}

__global__ __launch_bounds__(256) void fc_kernel(
    const float* __restrict__ wf1, const float* __restrict__ bf1,
    const float* __restrict__ wf2, const float* __restrict__ bf2,
    const unsigned* __restrict__ u, const int* __restrict__ max1p,
    const int* __restrict__ max2p, const int* __restrict__ pool,
    float* __restrict__ out) {
    __shared__ float red[256];
    __shared__ float pool_s[2048];
    __shared__ float y1s[4096];
    __shared__ float y2s[320];
    __shared__ float sh[4];
    int tid = threadIdx.x;

    float mm = 0.f;
    for (int i = tid; i < 8192; i += 256) mm = fmaxf(mm, fabsf(wf1[i]));
    red[tid] = mm; __syncthreads();
    for (int s = 128; s > 0; s >>= 1) { if (tid < s) red[tid] = fmaxf(red[tid], red[tid + s]); __syncthreads(); }
    if (tid == 0) sh[0] = fmaxf(red[0], 1e-8f) / 7.f;
    __syncthreads();
    mm = 0.f;
    for (int i = tid; i < 1280; i += 256) mm = fmaxf(mm, fabsf(wf2[i]));
    red[tid] = mm; __syncthreads();
    for (int s = 128; s > 0; s >>= 1) { if (tid < s) red[tid] = fmaxf(red[tid], red[tid + s]); __syncthreads(); }
    if (tid == 0) sh[1] = fmaxf(red[0], 1e-8f) / 7.f;
    __syncthreads();

    float swf1 = sh[0], swf2 = sh[1];
    float s_in = ws_scale(u, 0, 7.f);
    float sw1 = ws_scale(u, 1, 7.f);
    float sw2 = ws_scale(u, 2, 7.f);
    float s1 = fmaxf(s_in * sw1 * (float)(*max1p), 1e-8f) / 15.f;
    float s2 = fmaxf(s1 * sw2 * (float)(*max2p), 1e-8f) / 15.f;

    for (int i = tid; i < 2048; i += 256) pool_s[i] = s2 * (float)pool[i] * (1.f / 48400.f);
    __syncthreads();

    float sb1 = s2 * swf1;
    for (int idx = tid; idx < 4096; idx += 256) {
        int b = idx >> 7, n = idx & 127;
        float acc = rintf(bf1[n] / sb1) * sb1;
        for (int k = 0; k < 64; k++) {
            float wq = fminf(fmaxf(rintf(wf1[n * 64 + k] / swf1), -8.f), 7.f) * swf1;
            acc += pool_s[b * 64 + k] * wq;
        }
        y1s[idx] = fmaxf(acc, 0.f);
    }
    __syncthreads();

    mm = 0.f;
    for (int i = tid; i < 4096; i += 256) mm = fmaxf(mm, y1s[i]);
    red[tid] = mm; __syncthreads();
    for (int s = 128; s > 0; s >>= 1) { if (tid < s) red[tid] = fmaxf(red[tid], red[tid + s]); __syncthreads(); }
    if (tid == 0) sh[2] = fmaxf(red[0], 1e-8f) / 15.f;
    __syncthreads();
    float s3 = sh[2];
    for (int i = tid; i < 4096; i += 256)
        y1s[i] = fminf(fmaxf(rintf(y1s[i] / s3), 0.f), 15.f) * s3;
    __syncthreads();

    float sb2 = s3 * swf2;
    for (int idx = tid; idx < 320; idx += 256) {
        int b = idx / 10, j = idx % 10;
        float acc = rintf(bf2[j] / sb2) * sb2;
        for (int n = 0; n < 128; n++) {
            float wq = fminf(fmaxf(rintf(wf2[j * 128 + n] / swf2), -8.f), 7.f) * swf2;
            acc += y1s[b * 128 + n] * wq;
        }
        y2s[idx] = acc;
    }
    __syncthreads();

    if (tid < 32) {
        int b = tid;
        float mx = -1e30f;
        for (int j = 0; j < 10; j++) mx = fmaxf(mx, y2s[b * 10 + j]);
        float se = 0.f;
        for (int j = 0; j < 10; j++) se += __expf(y2s[b * 10 + j] - mx);
        float lse = __logf(se);
        for (int j = 0; j < 10; j++) out[b * 10 + j] = y2s[b * 10 + j] - mx - lse;
    }
}

extern "C" void kernel_launch(void* const* d_in, const int* in_sizes, int n_in,
                              void* d_out, int out_size, void* d_ws, size_t ws_size,
                              hipStream_t stream) {
    const float* x   = (const float*)d_in[0];
    const float* w1  = (const float*)d_in[1];
    const float* b1  = (const float*)d_in[2];
    const float* w2  = (const float*)d_in[3];
    const float* b2  = (const float*)d_in[4];
    const float* wf1 = (const float*)d_in[5];
    const float* bf1 = (const float*)d_in[6];
    const float* wf2 = (const float*)d_in[7];
    const float* bf2 = (const float*)d_in[8];
    float* out = (float*)d_out;

    char* ws = (char*)d_ws;
    unsigned* scal_u = (unsigned*)ws;
    int* max1p = (int*)ws + 3;
    int* max2p = (int*)ws + 4;
    int* pool  = (int*)(ws + OFF_POOL);
    int* w1p   = (int*)(ws + OFF_W1P);
    int* w2p   = (int*)(ws + OFF_W2P);
    signed char* xq = (signed char*)(ws + OFF_XQ);
    signed char* a1 = (signed char*)(ws + OFF_A1);

    zero_ws<<<9, 256, 0, stream>>>((int*)ws, 2304);

    const int NX = 32 * 3 * 224 * 224;
    absmax_kernel<<<1024, 256, 0, stream>>>(x, NX, scal_u + 0);
    absmax_kernel<<<4, 256, 0, stream>>>(w1, 864, scal_u + 1);
    absmax_kernel<<<72, 256, 0, stream>>>(w2, 18432, scal_u + 2);

    quantize_x<<<(NX + 255) / 256, 256, 0, stream>>>(x, scal_u, xq);
    pack_w1<<<2, 256, 0, stream>>>(w1, scal_u, w1p);
    pack_w2<<<20, 256, 0, stream>>>(w2, scal_u, w2p);

    dim3 g1(14, 222, 32);
    dim3 g2(14, 220, 32);
    conv1_wmma<<<g1, 32, 0, stream>>>(xq, w1p, b1, scal_u, max1p, a1, 0);
    conv1_wmma<<<g1, 32, 0, stream>>>(xq, w1p, b1, scal_u, max1p, a1, 1);
    conv2_wmma<<<g2, 32, 0, stream>>>(a1, w2p, b2, scal_u, max1p, max2p, pool, 0);
    conv2_wmma<<<g2, 32, 0, stream>>>(a1, w2p, b2, scal_u, max1p, max2p, pool, 1);

    fc_kernel<<<1, 256, 0, stream>>>(wf1, bf1, wf2, bf2, scal_u, max1p, max2p, pool, out);
}
